// MutilHeadSelfAttn_66013647340076
// MI455X (gfx1250) — matrix operand order repaired and checked by reading, not compiled, not run on hardware
//
#include <hip/hip_runtime.h>
#include <hip/hip_bf16.h>
#include <math.h>

// ---------------------------------------------------------------------------
// MI455X (gfx1250) fused multi-head self-attention block.
// Matmuls via v_wmma_f32_16x16x32_bf16; GEMM tiles staged by the Tensor Data
// Mover (tensor_load_to_lds + s_wait_tensorcnt), double-buffered in LDS.
// ---------------------------------------------------------------------------

typedef __bf16 bf16_t;
typedef __attribute__((ext_vector_type(4)))  __bf16 v4bf;
typedef __attribute__((ext_vector_type(8)))  __bf16 v8bf;
typedef __attribute__((ext_vector_type(16))) __bf16 v16bf;
typedef __attribute__((ext_vector_type(8)))  float  v8f;
typedef __attribute__((ext_vector_type(4)))  unsigned int v4u;
typedef __attribute__((ext_vector_type(8)))  unsigned int v8u;

#define NHEAD 16
#define HDIM  64
#define HID   1024
#define BSZ   4
#define SEQ   1024
#define NTOK  (BSZ*SEQ)

static __device__ __forceinline__ v16bf cat8(v8bf lo, v8bf hi) {
    return __builtin_shufflevector(lo, hi, 0,1,2,3,4,5,6,7,8,9,10,11,12,13,14,15);
}

// Max-reduce across each 16-lane half using DPP only (no LDS/bpermute).
// Schedule: xor1 (quad_perm), xor2 (quad_perm), xor7 (ROW_HALF_MIRROR),
// xor15 (ROW_MIRROR) — after each step every lane holds the max of its
// subgroup, so the non-power-of-two partners still cover the full 16 lanes.
static __device__ __forceinline__ float dpp_max16(float x) {
    int t;
    t = __builtin_amdgcn_update_dpp(0, __float_as_int(x), 0xB1,  0xf, 0xf, true); // quad_perm [1,0,3,2]
    x = fmaxf(x, __int_as_float(t));
    t = __builtin_amdgcn_update_dpp(0, __float_as_int(x), 0x4E,  0xf, 0xf, true); // quad_perm [2,3,0,1]
    x = fmaxf(x, __int_as_float(t));
    t = __builtin_amdgcn_update_dpp(0, __float_as_int(x), 0x141, 0xf, 0xf, true); // ROW_HALF_MIRROR
    x = fmaxf(x, __int_as_float(t));
    t = __builtin_amdgcn_update_dpp(0, __float_as_int(x), 0x140, 0xf, 0xf, true); // ROW_MIRROR
    x = fmaxf(x, __int_as_float(t));
    return x;
}

// ---------------- Tensor Data Mover: 2D bf16 tile -> LDS -------------------
// Builds the D# descriptor (CDNA5 ISA) and issues
// TENSOR_LOAD_TO_LDS. Issue once per wave; completion tracked by TENSORcnt.
static __device__ __forceinline__ void tdm_load_tile_bf16(
    const bf16_t* gsrc, unsigned int lds_byte_off,
    unsigned int tile_k, unsigned int tile_rows,
    unsigned int row_stride_elems, unsigned int tensor_rows)
{
    unsigned long long ga = (unsigned long long)(uintptr_t)gsrc;
    v4u g0;
    g0[0] = 1u;                                   // count=1, user descriptor
    g0[1] = lds_byte_off;                         // lds_addr
    g0[2] = (unsigned int)ga;                     // global_addr[31:0]
    g0[3] = (unsigned int)((ga >> 32) & 0x01FFFFFFu) | (2u << 30); // addr[56:32] | type=2
    const unsigned int td0 = row_stride_elems;    // tensor_dim0 (elements per row)
    const unsigned int td1 = tensor_rows;         // tensor_dim1 (rows)
    const unsigned int s0  = row_stride_elems;    // tensor_dim0_stride (elements)
    v8u g1;
    g1[0] = 1u << 16;                             // data_size=1 -> 2 bytes
    g1[1] = (td0 & 0xFFFFu) << 16;                // tensor_dim0[15:0]  @ bits 63:48
    g1[2] = (td0 >> 16) | ((td1 & 0xFFFFu) << 16);// tensor_dim0[31:16] | tensor_dim1[15:0]
    g1[3] = (td1 >> 16) | (tile_k << 16);         // tensor_dim1[31:16] | tile_dim0
    g1[4] = tile_rows;                            // tile_dim1 (tile_dim2 = 0)
    g1[5] = s0;                                   // tensor_dim0_stride[31:0]
    g1[6] = 0;                                    // stride[47:32]=0 | dim1_stride lo = 0
    g1[7] = 0;
    asm volatile("tensor_load_to_lds %0, %1"
                 :: "s"(g0), "s"(g1)
                 : "memory");
}

static __device__ __forceinline__ unsigned int lds_off_of(const void* p) {
    return (unsigned int)(uintptr_t)p;   // generic LDS pointer: low 32 bits = LDS offset
}

// --------------------------- RoPE table ------------------------------------
__global__ void rope_table_kernel(float* __restrict__ rc, float* __restrict__ rs) {
    int s = blockIdx.x;
    int d = threadIdx.x;          // 0..63, freq index = d % 32
    float fi  = (float)(d & 31);
    float inv = expf(fi * (-1.0f/32.0f) * 9.210340371976184f);  // 10000^(-fi/32)
    float ang = (float)s * inv;
    rc[s*HDIM + d] = cosf(ang);
    rs[s*HDIM + d] = sinf(ang);
}

// --------------------------- f32 -> bf16 -----------------------------------
__global__ __launch_bounds__(256) void cvt_bf16_kernel(const float* __restrict__ w,
                                                       bf16_t* __restrict__ o) {
    size_t i = ((size_t)blockIdx.x * 256 + threadIdx.x) * 4;
    float4 v = *(const float4*)(w + i);
    v4bf ob;
    ob[0] = (__bf16)v.x; ob[1] = (__bf16)v.y; ob[2] = (__bf16)v.z; ob[3] = (__bf16)v.w;
    *(v4bf*)(o + i) = ob;
}

// --------------------------- row LayerNorm ---------------------------------
__global__ __launch_bounds__(256) void ln_rows_kernel(const float* __restrict__ x,
                                                      const float* __restrict__ g,
                                                      const float* __restrict__ b,
                                                      bf16_t* __restrict__ out_bf,
                                                      float*  __restrict__ out_f32) {
    __shared__ float sh_s[8], sh_q[8], tot[2];
    const int row = blockIdx.x;
    const int tid = threadIdx.x;
    const float* xr = x + (size_t)row * HID;
    float4 v = *(const float4*)(xr + tid * 4);
    float s = v.x + v.y + v.z + v.w;
    float q = v.x*v.x + v.y*v.y + v.z*v.z + v.w*v.w;
    for (int off = 16; off > 0; off >>= 1) {
        s += __shfl_down(s, off, 32);
        q += __shfl_down(q, off, 32);
    }
    int wid = tid >> 5, lane = tid & 31;
    if (lane == 0) { sh_s[wid] = s; sh_q[wid] = q; }
    __syncthreads();
    if (tid == 0) {
        float ts = 0.f, tq = 0.f;
        for (int i = 0; i < 8; i++) { ts += sh_s[i]; tq += sh_q[i]; }
        tot[0] = ts; tot[1] = tq;
    }
    __syncthreads();
    float mu   = tot[0] * (1.0f/HID);
    float var  = tot[1] * (1.0f/HID) - mu*mu;
    float rstd = rsqrtf(var + 1e-5f);
    float4 gg = *(const float4*)(g + tid*4);
    float4 bb = *(const float4*)(b + tid*4);
    float4 y;
    y.x = (v.x - mu)*rstd*gg.x + bb.x;
    y.y = (v.y - mu)*rstd*gg.y + bb.y;
    y.z = (v.z - mu)*rstd*gg.z + bb.z;
    y.w = (v.w - mu)*rstd*gg.w + bb.w;
    v4bf ob;
    ob[0] = (__bf16)y.x; ob[1] = (__bf16)y.y; ob[2] = (__bf16)y.z; ob[3] = (__bf16)y.w;
    *(v4bf*)(out_bf + (size_t)row*HID + tid*4) = ob;
    if (out_f32) *(float4*)(out_f32 + (size_t)row*HID + tid*4) = y;
}

// --------------------------- final residual + LN ---------------------------
__global__ __launch_bounds__(256) void final_ln_add_kernel(const float* __restrict__ fx,
                                                           const float* __restrict__ fo,
                                                           const float* __restrict__ g,
                                                           const float* __restrict__ b,
                                                           float* __restrict__ out) {
    __shared__ float sh_s[8], sh_q[8], tot[2];
    const int row = blockIdx.x;
    const int tid = threadIdx.x;
    const float* fr = fo + (size_t)row * HID;
    float4 v = *(const float4*)(fr + tid*4);
    float s = v.x + v.y + v.z + v.w;
    float q = v.x*v.x + v.y*v.y + v.z*v.z + v.w*v.w;
    for (int off = 16; off > 0; off >>= 1) {
        s += __shfl_down(s, off, 32);
        q += __shfl_down(q, off, 32);
    }
    int wid = tid >> 5, lane = tid & 31;
    if (lane == 0) { sh_s[wid] = s; sh_q[wid] = q; }
    __syncthreads();
    if (tid == 0) {
        float ts = 0.f, tq = 0.f;
        for (int i = 0; i < 8; i++) { ts += sh_s[i]; tq += sh_q[i]; }
        tot[0] = ts; tot[1] = tq;
    }
    __syncthreads();
    float mu   = tot[0] * (1.0f/HID);
    float var  = tot[1] * (1.0f/HID) - mu*mu;
    float rstd = rsqrtf(var + 1e-5f);
    float4 gg = *(const float4*)(g + tid*4);
    float4 bb = *(const float4*)(b + tid*4);
    float4 fxv = *(const float4*)(fx + (size_t)row*HID + tid*4);
    float4 y;
    y.x = fxv.x + (v.x - mu)*rstd*gg.x + bb.x;
    y.y = fxv.y + (v.y - mu)*rstd*gg.y + bb.y;
    y.z = fxv.z + (v.z - mu)*rstd*gg.z + bb.z;
    y.w = fxv.w + (v.w - mu)*rstd*gg.w + bb.w;
    *(float4*)(out + (size_t)row*HID + tid*4) = y;
}

// --------------------------- WMMA GEMM  Y = A @ W^T + bias -----------------
// Tiles staged by the Tensor Data Mover, double-buffered in LDS; wave 0 issues
// the DMA and gates completion with s_wait_tensorcnt, the workgroup barrier
// publishes the tiles to the other 7 waves.
// Block tile 128(M) x 128(N), 8 waves, each wave 32x64 (8 WMMA tiles), K=32.
template<bool DO_ROPE, int EPI>
__global__ __launch_bounds__(256) void gemm_kernel(const bf16_t* __restrict__ A,
                                                   const bf16_t* __restrict__ W,
                                                   const float*  __restrict__ bias,
                                                   const float*  __restrict__ rc,
                                                   const float*  __restrict__ rs,
                                                   bf16_t* __restrict__ outb,
                                                   float*  __restrict__ outf) {
    __shared__ __align__(128) bf16_t As[2][128*32];
    __shared__ __align__(128) bf16_t Bs[2][128*32];
    const int tid  = threadIdx.x;
    const int wid  = tid >> 5, lane = tid & 31;
    const int wr   = wid & 3,  wc   = wid >> 2;
    const int lh   = lane >> 4, ln  = lane & 15;
    const int m0   = blockIdx.y * 128;
    const int n0   = blockIdx.x * 128;

    v8f acc[2][4];
    for (int ah = 0; ah < 2; ah++)
        for (int t = 0; t < 4; t++)
            for (int r = 0; r < 8; r++) acc[ah][t][r] = 0.f;

    const bf16_t* Abase = A + (size_t)m0 * HID;
    const bf16_t* Wbase = W + (size_t)n0 * HID;

    if (wid == 0) {   // prime buffer 0 via TDM
        tdm_load_tile_bf16(Abase, lds_off_of(&As[0][0]), 32, 128, HID, NTOK);
        tdm_load_tile_bf16(Wbase, lds_off_of(&Bs[0][0]), 32, 128, HID, HID);
    }

    for (int kk = 0; kk < HID; kk += 32) {
        const int cur = (kk >> 5) & 1;
        if (wid == 0) {
            if (kk + 32 < HID) {   // prefetch next tiles into the other buffer
                tdm_load_tile_bf16(Abase + kk + 32, lds_off_of(&As[cur^1][0]), 32, 128, HID, NTOK);
                tdm_load_tile_bf16(Wbase + kk + 32, lds_off_of(&Bs[cur^1][0]), 32, 128, HID, HID);
                __builtin_amdgcn_s_wait_tensorcnt(2);  // in-order: oldest 2 (cur buf) done
            } else {
                __builtin_amdgcn_s_wait_tensorcnt(0);
            }
        }
        __syncthreads();

        // two A fragments (rows wr*32 .. +15 and +16 .. +31)
        const bf16_t* ap0 = &As[cur][(wr*32 + ln)*32 + lh*8];
        const bf16_t* ap1 = ap0 + 16*32;
        v16bf a0 = cat8(*(const v8bf*)ap0, *(const v8bf*)(ap0 + 16));
        v16bf a1 = cat8(*(const v8bf*)ap1, *(const v8bf*)(ap1 + 16));
        #pragma unroll
        for (int t = 0; t < 4; t++) {
            // B fragment: lane ln = col n, k at lh*8 .. +7 and 16+lh*8 .. +7
            const bf16_t* bp = &Bs[cur][(wc*64 + t*16 + ln)*32 + lh*8];
            v16bf bfr = cat8(*(const v8bf*)bp, *(const v8bf*)(bp + 16));
            acc[0][t] = __builtin_amdgcn_wmma_f32_16x16x32_bf16(
                false, a0, false, bfr, (short)0, acc[0][t], false, false);
            acc[1][t] = __builtin_amdgcn_wmma_f32_16x16x32_bf16(
                false, a1, false, bfr, (short)0, acc[1][t], false, false);
        }
        __syncthreads();   // all waves done with 'cur' before it is refilled
    }

    const int head = blockIdx.x*2 + wc;   // wave column == one head (64 feats)
    #pragma unroll
    for (int ah = 0; ah < 2; ah++) {
        const int mbase = m0 + wr*32 + ah*16 + lh*8;  // C layout: lanes>=16 -> +8
        if (EPI == 1) {
            #pragma unroll
            for (int t = 0; t < 4; t++) {
                int n = n0 + wc*64 + t*16 + ln;
                float bn = bias[n];
                #pragma unroll
                for (int r = 0; r < 8; r++) {
                    int m = mbase + r;
                    outf[(size_t)m*HID + n] = fmaxf(acc[ah][t][r] + bn, 0.f);
                }
            }
        } else if (DO_ROPE) {
            #pragma unroll
            for (int t = 0; t < 2; t++) {
                int d1 = t*16 + ln, d2 = d1 + 32;
                float b1v = bias[head*64 + d1], b2v = bias[head*64 + d2];
                #pragma unroll
                for (int r = 0; r < 8; r++) {
                    int m = mbase + r;
                    int bb = m >> 10, s = m & 1023;
                    float y1 = acc[ah][t][r]   + b1v;
                    float y2 = acc[ah][t+2][r] + b2v;
                    float c1 = rc[s*HDIM + d1], s1 = rs[s*HDIM + d1];
                    float c2 = rc[s*HDIM + d2], s2 = rs[s*HDIM + d2];
                    size_t base = (((size_t)bb*NHEAD + head)*SEQ + s)*HDIM;
                    outb[base + d1] = (__bf16)(y1*c1 - y2*s1);
                    outb[base + d2] = (__bf16)(y2*c2 + y1*s2);
                }
            }
        } else {
            #pragma unroll
            for (int t = 0; t < 4; t++) {
                int d = t*16 + ln;
                float bn = bias[head*64 + d];
                #pragma unroll
                for (int r = 0; r < 8; r++) {
                    int m = mbase + r;
                    int bb = m >> 10, s = m & 1023;
                    outb[(((size_t)bb*NHEAD + head)*SEQ + s)*HDIM + d] =
                        (__bf16)(acc[ah][t][r] + bn);
                }
            }
        }
    }
}

// --------------------------- flash attention -------------------------------
// grid (SEQ/128, NHEAD, BSZ), 256 threads = 8 waves, each wave owns 16 q rows.
// Online softmax. Row max reduced with DPP; row sum computed with an extra
// WMMA against an all-ones B fragment (denominator rides in accsum with the
// same rescale factor as the output accumulators).
__global__ __launch_bounds__(256) void attn_kernel(const bf16_t* __restrict__ Q,
                                                   const bf16_t* __restrict__ K,
                                                   const bf16_t* __restrict__ V,
                                                   const int*    __restrict__ mask,
                                                   float* __restrict__ X) {
    __shared__ __align__(64) bf16_t Ks[32*64];        // [key][d]
    __shared__ __align__(64) bf16_t Vs[64*32];        // transposed: [d][key]
    __shared__ __align__(64) bf16_t Ps[8][16*32];     // per-wave P relayout scratch
    const int tid  = threadIdx.x, wid = tid >> 5, lane = tid & 31;
    const int lh   = lane >> 4,   ln  = lane & 15;
    const int b    = blockIdx.z,  h   = blockIdx.y;
    const int q0   = blockIdx.x*128 + wid*16;
    const size_t bh = ((size_t)b*NHEAD + h)*SEQ;

    // this wave's Q fragments (d 0..31 and 32..63) in A layout
    const bf16_t* qrow = Q + (bh + q0 + ln)*HDIM;
    const int kb = lh*8;
    v16bf aq0 = cat8(*(const v8bf*)(qrow + kb),      *(const v8bf*)(qrow + kb + 16));
    v16bf aq1 = cat8(*(const v8bf*)(qrow + 32 + kb), *(const v8bf*)(qrow + 48 + kb));

    v16bf bones;
    #pragma unroll
    for (int i = 0; i < 16; i++) bones[i] = (__bf16)1.0f;

    v8f acc[4], accsum;
    float mrow[8];
    for (int t = 0; t < 4; t++)
        for (int r = 0; r < 8; r++) acc[t][r] = 0.f;
    for (int r = 0; r < 8; r++) { accsum[r] = 0.f; mrow[r] = -INFINITY; }

    const int skey = tid >> 3;         // 0..31
    const int sdc  = (tid & 7) * 8;    // 0..56
    const int* mrow_ptr = mask + b*SEQ;

    for (int kt = 0; kt < SEQ; kt += 32) {
        __syncthreads();               // protect LDS tiles from previous iter
        // stage K tile [32 keys][64 d]
        *(v8bf*)&Ks[skey*64 + sdc] = *(const v8bf*)(K + (bh + kt + skey)*HDIM + sdc);
        // stage V tile transposed [d][key]
        v8bf vv = *(const v8bf*)(V + (bh + kt + skey)*HDIM + sdc);
        #pragma unroll
        for (int i = 0; i < 8; i++) Vs[(sdc + i)*32 + skey] = vv[i];
        if (kt + 32 < SEQ) {
            __builtin_prefetch(K + (bh + kt + 32 + skey)*HDIM + sdc, 0, 3);
            __builtin_prefetch(V + (bh + kt + 32 + skey)*HDIM + sdc, 0, 3);
        }
        __syncthreads();

        // scores for key halves f=0,1: two WMMAs each (d halves)
        v8f sc[2];
        #pragma unroll
        for (int f = 0; f < 2; f++) {
            v8f c;
            for (int r = 0; r < 8; r++) c[r] = 0.f;
            v16bf bk0 = cat8(*(const v8bf*)&Ks[(f*16 + ln)*64 + lh*8], *(const v8bf*)&Ks[(f*16 + ln)*64 + 16 + lh*8]);
            v16bf bk1 = cat8(*(const v8bf*)&Ks[(f*16 + ln)*64 + 32 + lh*8], *(const v8bf*)&Ks[(f*16 + ln)*64 + 48 + lh*8]);
            c = __builtin_amdgcn_wmma_f32_16x16x32_bf16(false, aq0, false, bk0, (short)0, c, false, false);
            c = __builtin_amdgcn_wmma_f32_16x16x32_bf16(false, aq1, false, bk1, (short)0, c, false, false);
            int mk = mrow_ptr[kt + f*16 + ln];
            #pragma unroll
            for (int r = 0; r < 8; r++)
                sc[f][r] = mk ? c[r]*0.125f : 1e-10f;
        }

        // online softmax: DPP row-max, exp, rescale factor
        float fac[8];
        #pragma unroll
        for (int r = 0; r < 8; r++) {
            float tm = dpp_max16(fmaxf(sc[0][r], sc[1][r]));
            float mn = fmaxf(mrow[r], tm);
            fac[r] = __expf(mrow[r] - mn);
            mrow[r] = mn;
            sc[0][r] = __expf(sc[0][r] - mn);
            sc[1][r] = __expf(sc[1][r] - mn);
        }

        // rescale accumulators (incl. denominator); dump P to LDS scratch
        #pragma unroll
        for (int r = 0; r < 8; r++) {
            #pragma unroll
            for (int t = 0; t < 4; t++) acc[t][r] *= fac[r];
            accsum[r] *= fac[r];
            Ps[wid][(lh*8 + r)*32 + ln]      = (__bf16)sc[0][r];
            Ps[wid][(lh*8 + r)*32 + 16 + ln] = (__bf16)sc[1][r];
        }
        __syncthreads();

        // reload P in A layout; P@V plus P@ones (row sums)
        const bf16_t* pp = &Ps[wid][ln*32 + lh*8];
        v16bf pa = cat8(*(const v8bf*)pp, *(const v8bf*)(pp + 16));
        #pragma unroll
        for (int t = 0; t < 4; t++) {
            v16bf bv = cat8(*(const v8bf*)&Vs[(t*16 + ln)*32 + lh*8], *(const v8bf*)&Vs[(t*16 + ln)*32 + 16 + lh*8]);
            acc[t] = __builtin_amdgcn_wmma_f32_16x16x32_bf16(
                false, pa, false, bv, (short)0, acc[t], false, false);
        }
        accsum = __builtin_amdgcn_wmma_f32_16x16x32_bf16(
            false, pa, false, bones, (short)0, accsum, false, false);
    }

    // normalize and write x[b][s][h*64+d] (fp32, feeds LayerNorm)
    #pragma unroll
    for (int r = 0; r < 8; r++) {
        float inv = 1.0f / accsum[r];
        int s = q0 + lh*8 + r;
        #pragma unroll
        for (int t = 0; t < 4; t++) {
            int d = t*16 + ln;
            X[((size_t)b*SEQ + s)*HID + h*HDIM + d] = acc[t][r]*inv;
        }
    }
}

// ---------------------------------------------------------------------------
extern "C" void kernel_launch(void* const* d_in, const int* in_sizes, int n_in,
                              void* d_out, int out_size, void* d_ws, size_t ws_size,
                              hipStream_t stream) {
    (void)in_sizes; (void)n_in; (void)out_size; (void)ws_size;
    const float* q    = (const float*)d_in[0];
    const float* k    = (const float*)d_in[1];
    const float* v    = (const float*)d_in[2];
    const float* g1q  = (const float*)d_in[3];
    const float* b1q  = (const float*)d_in[4];
    const float* g1k  = (const float*)d_in[5];
    const float* b1k  = (const float*)d_in[6];
    const float* g1v  = (const float*)d_in[7];
    const float* b1v  = (const float*)d_in[8];
    const float* Wq   = (const float*)d_in[9];
    const float* bq   = (const float*)d_in[10];
    const float* Wk   = (const float*)d_in[11];
    const float* bk   = (const float*)d_in[12];
    const float* Wv   = (const float*)d_in[13];
    const float* bv   = (const float*)d_in[14];
    const float* g2   = (const float*)d_in[15];
    const float* b2   = (const float*)d_in[16];
    const float* g3   = (const float*)d_in[17];
    const float* b3   = (const float*)d_in[18];
    const float* Wf   = (const float*)d_in[19];
    const float* bfb  = (const float*)d_in[20];
    const int*   mask = (const int*)  d_in[21];
    float* out = (float*)d_out;

    char* ws = (char*)d_ws;
    size_t off = 0;
    auto alloc = [&](size_t bytes) -> char* {
        char* p = ws + off;
        off += (bytes + 255) & ~(size_t)255;
        return p;
    };
    const size_t WBYTES = (size_t)HID*HID*sizeof(bf16_t);            // 2 MB
    const size_t ABYTES = (size_t)NTOK*HID*sizeof(bf16_t);           // 8 MB
    const size_t QBYTES = (size_t)BSZ*NHEAD*SEQ*HDIM*sizeof(bf16_t); // 8 MB
    bf16_t* WqB = (bf16_t*)alloc(WBYTES);
    bf16_t* WkB = (bf16_t*)alloc(WBYTES);
    bf16_t* WvB = (bf16_t*)alloc(WBYTES);
    bf16_t* WfB = (bf16_t*)alloc(WBYTES);
    bf16_t* AqB = (bf16_t*)alloc(ABYTES);
    bf16_t* AkB = (bf16_t*)alloc(ABYTES);
    bf16_t* AvB = (bf16_t*)alloc(ABYTES);
    float*  rc  = (float*) alloc((size_t)SEQ*HDIM*sizeof(float));
    float*  rs  = (float*) alloc((size_t)SEQ*HDIM*sizeof(float));
    bf16_t* Qbf = (bf16_t*)alloc(QBYTES);
    bf16_t* Kbf = (bf16_t*)alloc(QBYTES);
    bf16_t* Vbf = (bf16_t*)alloc(QBYTES);
    float*  X   = (float*) alloc((size_t)NTOK*HID*sizeof(float));
    float*  fxF = (float*) alloc((size_t)NTOK*HID*sizeof(float));
    bf16_t* fxB = (bf16_t*)alloc(ABYTES);
    float*  foF = (float*) alloc((size_t)NTOK*HID*sizeof(float));

    // 1) RoPE tables + bf16 weights
    rope_table_kernel<<<dim3(SEQ), dim3(HDIM), 0, stream>>>(rc, rs);
    cvt_bf16_kernel<<<dim3(HID*HID/1024), 256, 0, stream>>>(Wq, WqB);
    cvt_bf16_kernel<<<dim3(HID*HID/1024), 256, 0, stream>>>(Wk, WkB);
    cvt_bf16_kernel<<<dim3(HID*HID/1024), 256, 0, stream>>>(Wv, WvB);
    cvt_bf16_kernel<<<dim3(HID*HID/1024), 256, 0, stream>>>(Wf, WfB);

    // 2) input LayerNorms -> bf16 activations
    ln_rows_kernel<<<NTOK, 256, 0, stream>>>(q, g1q, b1q, AqB, nullptr);
    ln_rows_kernel<<<NTOK, 256, 0, stream>>>(k, g1k, b1k, AkB, nullptr);
    ln_rows_kernel<<<NTOK, 256, 0, stream>>>(v, g1v, b1v, AvB, nullptr);

    // 3) QKV projections (WMMA + TDM staging), fused RoPE for Q/K
    dim3 ggrid(HID/128, NTOK/128);
    gemm_kernel<true, 0><<<ggrid, 256, 0, stream>>>(AqB, WqB, bq, rc, rs, Qbf, nullptr);
    gemm_kernel<true, 0><<<ggrid, 256, 0, stream>>>(AkB, WkB, bk, rc, rs, Kbf, nullptr);
    gemm_kernel<false,0><<<ggrid, 256, 0, stream>>>(AvB, WvB, bv, rc, rs, Vbf, nullptr);

    // 4) flash attention
    attn_kernel<<<dim3(SEQ/128, NHEAD, BSZ), 256, 0, stream>>>(Qbf, Kbf, Vbf, mask, X);

    // 5) fx = LN(x) (fp32 for residual + bf16 for GEMM)
    ln_rows_kernel<<<NTOK, 256, 0, stream>>>(X, g2, b2, fxB, fxF);

    // 6) fo = relu(fx @ Wf^T + bf)
    gemm_kernel<false,1><<<ggrid, 256, 0, stream>>>(fxB, WfB, bfb, nullptr, nullptr, nullptr, foF);

    // 7) out = fx + LN(fo)
    final_ln_add_kernel<<<NTOK, 256, 0, stream>>>(fxF, foF, g3, b3, out);
}
